// Int8Linear_78443282694184
// MI455X (gfx1250) — 8-bit matrix operand order repaired and checked by reading, compiled, not run on hardware
//
#include <hip/hip_runtime.h>

typedef __attribute__((ext_vector_type(8))) int   v8i;
typedef __attribute__((ext_vector_type(4))) int   v4i;
typedef __attribute__((ext_vector_type(2))) int   v2i;
typedef __attribute__((ext_vector_type(4))) float v4f;

#define DIN   4096
#define DOUT  4096
#define MROWS 8192

#if __has_builtin(__builtin_amdgcn_global_load_async_to_lds_b128) && \
    __has_builtin(__builtin_amdgcn_s_wait_asynccnt)
#define HAVE_ASYNC_LDS 1
#else
#define HAVE_ASYNC_LDS 0
#endif

#if HAVE_ASYNC_LDS
typedef __attribute__((address_space(1))) v4i* gv4p;
typedef __attribute__((address_space(3))) v4i* lv4p;
#define GLD_ASYNC_B128(g, l)                                              \
  __builtin_amdgcn_global_load_async_to_lds_b128(                         \
      (gv4p)(v4i*)(g), (lv4p)(v4i*)(l), 0, 0)
#endif

__global__ __launch_bounds__(256) void quant_x_kernel(
    const float* __restrict__ x, const float* __restrict__ act_scale,
    unsigned int* __restrict__ q4, int n4) {
  int i = blockIdx.x * blockDim.x + threadIdx.x;
  if (i >= n4) return;
  const float inv = 1.0f / act_scale[0];
  v4f v = ((const v4f*)x)[i];
  unsigned int b0 = (unsigned int)(int)rintf(fminf(fmaxf(v.x * inv, 0.0f), 255.0f));
  unsigned int b1 = (unsigned int)(int)rintf(fminf(fmaxf(v.y * inv, 0.0f), 255.0f));
  unsigned int b2 = (unsigned int)(int)rintf(fminf(fmaxf(v.z * inv, 0.0f), 255.0f));
  unsigned int b3 = (unsigned int)(int)rintf(fminf(fmaxf(v.w * inv, 0.0f), 255.0f));
  q4[i] = b0 | (b1 << 8) | (b2 << 16) | (b3 << 24);
}

__global__ __launch_bounds__(256) void pack_w_kernel(
    const int* __restrict__ w, unsigned int* __restrict__ w8_4, int n4) {
  int i = blockIdx.x * blockDim.x + threadIdx.x;
  if (i >= n4) return;
  v4i v = ((const v4i*)w)[i];
  w8_4[i] = (v.x & 0xff) | ((v.y & 0xff) << 8) | ((v.z & 0xff) << 16)
          | ((v.w & 0xff) << 24);
}

__device__ __forceinline__ void compute_chunk(
    const unsigned char* lA, const unsigned char* lB,
    int wm, int wn, int nl, int hi, v8i (&acc)[4][2]) {
  v8i afrag[4];
#pragma unroll
  for (int mi = 0; mi < 4; ++mi) {
    const unsigned char* ap = lA + (wm * 64 + mi * 16 + nl) * 64 + hi * 16;
    v2i p0 = *(const v2i*)(ap);
    v2i p1 = *(const v2i*)(ap + 8);
    v2i p2 = *(const v2i*)(ap + 32);
    v2i p3 = *(const v2i*)(ap + 40);
    v8i a = {p0.x, p0.y, p1.x, p1.y, p2.x, p2.y, p3.x, p3.y};
    afrag[mi] = a;
  }
  v8i bfrag[2];
#pragma unroll
  for (int ni = 0; ni < 2; ++ni) {
    const unsigned char* bp = lB + (wn * 32 + ni * 16 + nl) * 64 + hi * 16;
    v4i lo  = *(const v4i*)(bp);
    v4i hi4 = *(const v4i*)(bp + 32);
    v8i b = {lo.x, lo.y, lo.z, lo.w, hi4.x, hi4.y, hi4.z, hi4.w};
    bfrag[ni] = b;
  }
#pragma unroll
  for (int mi = 0; mi < 4; ++mi)
#pragma unroll
    for (int ni = 0; ni < 2; ++ni)
      acc[mi][ni] = __builtin_amdgcn_wmma_i32_16x16x64_iu8(
           false, afrag[mi],
           true,  bfrag[ni],
          acc[mi][ni],  false,  false);
}

__global__ __launch_bounds__(256) void int8_gemm_wmma(
    const unsigned char* __restrict__ q, const signed char* __restrict__ w8,
    const float* __restrict__ wscale, const float* __restrict__ bias,
    const float* __restrict__ act_scale, float* __restrict__ out) {
  __shared__ __align__(16) unsigned char As[2][128 * 64];
  __shared__ __align__(16) unsigned char Bs[2][128 * 64];

  const int tid  = threadIdx.x;
  const int lane = tid & 31;
  const int wave = tid >> 5;
  const int wm   = wave & 1;
  const int wn   = wave >> 1;

  const int mBase = blockIdx.y * 128;
  const int nBase = blockIdx.x * 128;

  const int srow = tid >> 2;
  const int scol = (tid & 3) * 16;

  const int nl = lane & 15;
  const int hi = lane >> 4;

  const unsigned char* gA = q + (size_t)mBase * DIN;
  const unsigned char* gB = (const unsigned char*)w8 + (size_t)nBase * DIN;
  const size_t rowA0 = (size_t)srow * DIN + scol;
  const size_t rowA1 = (size_t)(srow + 64) * DIN + scol;
  const int    lrow0 = srow * 64 + scol;
  const int    lrow1 = (srow + 64) * 64 + scol;

  v8i acc[4][2];
#pragma unroll
  for (int mi = 0; mi < 4; ++mi)
#pragma unroll
    for (int ni = 0; ni < 2; ++ni)
#pragma unroll
      for (int r = 0; r < 8; ++r) acc[mi][ni][r] = 0;

  const int NT = DIN / 64;

#if HAVE_ASYNC_LDS
  unsigned char* curA = As[0];
  unsigned char* nxtA = As[1];
  unsigned char* curB = Bs[0];
  unsigned char* nxtB = Bs[1];

  GLD_ASYNC_B128(gA + rowA0, curA + lrow0);
  GLD_ASYNC_B128(gA + rowA1, curA + lrow1);
  GLD_ASYNC_B128(gB + rowA0, curB + lrow0);
  GLD_ASYNC_B128(gB + rowA1, curB + lrow1);

  const unsigned char* pA0 = gA + rowA0 + 64;
  const unsigned char* pA1 = gA + rowA1 + 64;
  const unsigned char* pB0 = gB + rowA0 + 64;
  const unsigned char* pB1 = gB + rowA1 + 64;

#pragma clang loop unroll(disable)
  for (int t = 0; t < NT - 1; ++t) {
    GLD_ASYNC_B128(pA0, nxtA + lrow0);
    GLD_ASYNC_B128(pA1, nxtA + lrow1);
    GLD_ASYNC_B128(pB0, nxtB + lrow0);
    GLD_ASYNC_B128(pB1, nxtB + lrow1);
    pA0 += 64; pA1 += 64; pB0 += 64; pB1 += 64;
    __builtin_amdgcn_s_wait_asynccnt(4);
    __syncthreads();
    compute_chunk(curA, curB, wm, wn, nl, hi, acc);
    __syncthreads();
    unsigned char* tA = curA; curA = nxtA; nxtA = tA;
    unsigned char* tB = curB; curB = nxtB; nxtB = tB;
  }
  __builtin_amdgcn_s_wait_asynccnt(0);
  __syncthreads();
  compute_chunk(curA, curB, wm, wn, nl, hi, acc);
#else
  for (int t = 0; t < NT; ++t) {
    const int kc = t * 64;
    v4i a0 = *(const v4i*)(gA + rowA0 + kc);
    v4i a1 = *(const v4i*)(gA + rowA1 + kc);
    v4i b0 = *(const v4i*)(gB + rowA0 + kc);
    v4i b1 = *(const v4i*)(gB + rowA1 + kc);
    __syncthreads();
    *(v4i*)(As[0] + lrow0) = a0;
    *(v4i*)(As[0] + lrow1) = a1;
    *(v4i*)(Bs[0] + lrow0) = b0;
    *(v4i*)(Bs[0] + lrow1) = b1;
    __syncthreads();
    compute_chunk(As[0], Bs[0], wm, wn, nl, hi, acc);
  }
#endif

  const float s = act_scale[0];
#pragma unroll
  for (int ni = 0; ni < 2; ++ni) {
    const int ncol = nBase + wn * 32 + ni * 16 + nl;
    const float sc = s * wscale[ncol];
    const float bo = bias[ncol];
#pragma unroll
    for (int mi = 0; mi < 4; ++mi) {
      const int mrow = mBase + wm * 64 + mi * 16 + hi * 8;
#pragma unroll
      for (int r = 0; r < 8; ++r)
        out[(size_t)(mrow + r) * DOUT + ncol] =
            (float)acc[mi][ni][r] * sc + bo;
    }
  }
}

extern "C" void kernel_launch(void* const* d_in, const int* in_sizes, int n_in,
                              void* d_out, int out_size, void* d_ws, size_t ws_size,
                              hipStream_t stream) {
  const float* x         = (const float*)d_in[0];
  const int*   w         = (const int*)d_in[1];
  const float* wscale    = (const float*)d_in[2];
  const float* bias      = (const float*)d_in[3];
  const float* act_scale = (const float*)d_in[4];
  float*       out       = (float*)d_out;

  unsigned char* q  = (unsigned char*)d_ws;
  signed char*   w8 = (signed char*)d_ws + (size_t)MROWS * DIN;

  const int nx4 = MROWS * DIN / 4;
  quant_x_kernel<<<nx4 / 256, 256, 0, stream>>>(x, act_scale,
                                                (unsigned int*)q, nx4);
  const int nw4 = DOUT * DIN / 4;
  pack_w_kernel<<<nw4 / 256, 256, 0, stream>>>(w, (unsigned int*)w8, nw4);

  dim3 grid(DOUT / 128, MROWS / 128);
  int8_gemm_wmma<<<grid, 256, 0, stream>>>(q, w8, wscale, bias, act_scale, out);
}
